// GridLSTM_69449621176506
// MI455X (gfx1250) — compile-verified, matrix operand order checked by reading, not run on hardware
//
#include <hip/hip_runtime.h>
#include <hip/hip_bf16.h>
#include <stddef.h>
#include <stdint.h>

// Grid-LSTM on MI455X (gfx1250): wavefront over anti-diagonals, one launch per
// diagonal. Per cell one fused bf16 GEMM (M=128, N=1024, K=384) via
// v_wmma_f32_16x16x32_bf16. Both the A panel (x|h_col|h_row) and the B weight
// panels are staged into padded LDS by the Tensor Data Mover; B is
// double-buffered so TDM streams the next unit-tile under the current WMMAs.
// LSTM elementwise is fused in registers.

typedef __attribute__((ext_vector_type(16))) __bf16       v16bf;
typedef __attribute__((ext_vector_type(8)))  float        v8f;
typedef __attribute__((ext_vector_type(4)))  unsigned int u32x4;
typedef __attribute__((ext_vector_type(8)))  int          i32x8;
typedef __attribute__((ext_vector_type(4)))  int          i32x4;

#define BATCHN     128
#define HDIM       128
#define KTOT       384                  // 128 (x) + 128 (h_col) + 128 (h_row)
#define REG_STRIDE 136                  // bf16 per 128-elem chunk incl. 8 pad
#define REG_ELEMS  (128 * REG_STRIDE)   // A region: 128 rows
#define A_ELEMS    (3 * REG_ELEMS)      // 52224 bf16 = 104448 B
#define BCHUNKS    (64 * 3)             // B tile: 64 rows x 3 chunks
#define BBUF_ELEMS (BCHUNKS * REG_STRIDE) // 26112 bf16 = 52224 B
#define SLOTE      (32 * BATCHN * HDIM) // one parity slot of a state array

#if __has_builtin(__builtin_amdgcn_tensor_load_to_lds)
#define USE_TDM 1
#else
#define USE_TDM 0
#endif

__device__ __forceinline__ float sigmoidf_(float x) {
    return 1.0f / (1.0f + __expf(-x));
}

#if USE_TDM
// 2D TDM descriptor: `rows` sub-rows of 128 bf16 from a contiguous source,
// LDS destination padded +16B per 256B sub-row (-> 272B chunk stride).
__device__ __forceinline__ void tdm_load_padded(unsigned lds_off, const __bf16* src, int rows)
{
    unsigned long long ga = (unsigned long long)(uintptr_t)src;
    u32x4 g0;
    g0.x = 1u;                                            // count=1 (valid D#)
    g0.y = lds_off;                                       // lds_addr (bytes)
    g0.z = (unsigned)ga;                                  // global_addr[31:0]
    g0.w = (unsigned)((ga >> 32) & 0x1FFFFFFull)          // global_addr[56:32]
         | (2u << 30);                                    // type = 2 ("image")
    i32x8 g1;
    g1[0] = (1 << 16)                                     // data_size = 2B
          | (1 << 20)                                     // pad_enable
          | (5 << 22)                                     // pad_interval: 64 DW (256B)
          | (3 << 25);                                    // pad_amount: 4 DW (16B)
    g1[1] = (128 << 16);                                  // tensor_dim0 lo16 = 128
    g1[2] = (rows << 16);                                 // dim0 hi=0 | tensor_dim1 lo16
    g1[3] = (128 << 16);                                  // dim1 hi=0 | tile_dim0 = 128
    g1[4] = rows & 0xFFFF;                                // tile_dim1 | tile_dim2 = 0
    g1[5] = 128;                                          // tensor_dim0_stride lo32
    g1[6] = 0;
    g1[7] = 0;
    i32x4 z4 = {0, 0, 0, 0};
#if __clang_major__ >= 23
    i32x8 z8 = {0, 0, 0, 0, 0, 0, 0, 0};
    __builtin_amdgcn_tensor_load_to_lds(g0, g1, z4, z4, z8, 0);
#else
    __builtin_amdgcn_tensor_load_to_lds(g0, g1, z4, z4, 0);
#endif
}
#endif

// ---------------------------------------------------------------- prep ------
// Pack weights to bf16 tile-major [t=unit>>4][gate][unit&15][K], fuse biases,
// convert x to bf16, build a zero tile.
__global__ __launch_bounds__(256) void prep_k(
    const float* __restrict__ x,
    const float* __restrict__ w_ih, const float* __restrict__ w_hh,
    const float* __restrict__ b_ih, const float* __restrict__ b_hh,
    __bf16* __restrict__ Wb, float* __restrict__ bias,
    __bf16* __restrict__ xb, __bf16* __restrict__ zeroB)
{
    int gid = blockIdx.x * 256 + threadIdx.x;
    if (gid < 32 * 32 * BATCHN * 128) xb[gid] = (__bf16)x[gid];
    if (gid < 1024 * KTOT) {
        int j = gid / KTOT;                 // original gate row: g*256 + u
        int k = gid - j * KTOT;
        int g = j >> 8, u = j & 255;
        float w = (k < 128) ? w_ih[j * 128 + k] : w_hh[j * 256 + (k - 128)];
        Wb[(size_t)((((u >> 4) * 4 + g) * 16) + (u & 15)) * KTOT + k] = (__bf16)w;
    }
    if (gid < BATCHN * HDIM) zeroB[gid] = (__bf16)0.0f;
    if (gid < 1024) bias[gid] = b_ih[gid] + b_hh[gid];
}

// ------------------------------------------------------------ cell/diag -----
__global__ __launch_bounds__(256) void grid_lstm_diag(
    const __bf16* __restrict__ xb, const __bf16* __restrict__ Wb,
    const float* __restrict__ bias, const __bf16* __restrict__ zeroB,
    const __bf16* __restrict__ hColR, __bf16* __restrict__ hColW,
    const float*  __restrict__ cColR, float*  __restrict__ cColW,
    const __bf16* __restrict__ hRowR, __bf16* __restrict__ hRowW,
    const float*  __restrict__ cRowR, float*  __restrict__ cRowW,
    int d)
{
    extern __shared__ __align__(16) char smem_raw[];
    __bf16* ldsA = (__bf16*)smem_raw;          // 3 regions of [128][REG_STRIDE]
    __bf16* ldsB = ldsA + A_ELEMS;             // 2 x BBUF_ELEMS (double buffer)

    const int rmin      = (d > 31) ? (d - 31) : 0;
    const int cellIdx   = blockIdx.x >> 2;
    const int blkInCell = blockIdx.x & 3;
    const int r   = rmin + cellIdx;
    const int c   = d - r;
    const int tid = threadIdx.x;

    const __bf16* srcX = xb + (size_t)(r * 32 + c) * (BATCHN * 128);
    const __bf16* srcH = (c == 0) ? zeroB : (hColR + r * (BATCHN * HDIM));
    const __bf16* srcV = (r == 0) ? zeroB : (hRowR + c * (BATCHN * HDIM));
    const __bf16* srcB = Wb + (size_t)(blkInCell * 4) * (64 * KTOT);  // + ut*64*KTOT

    // ---- prologue staging: A panel (3 regions) + B tile 0 ------------------
#if USE_TDM
    const unsigned ldsABase = (unsigned)(uintptr_t)ldsA;
    const unsigned ldsBBase = (unsigned)(uintptr_t)ldsB;
    if (tid < 32) {                            // wave 0 drives the TDM
        tdm_load_padded(ldsABase + 0u * (REG_ELEMS * 2), srcX, 128);
        tdm_load_padded(ldsABase + 1u * (REG_ELEMS * 2), srcH, 128);
        tdm_load_padded(ldsABase + 2u * (REG_ELEMS * 2), srcV, 128);
        tdm_load_padded(ldsBBase, srcB, BCHUNKS);
    }
#else
    {
        const __bf16* srcs[3] = { srcX, srcH, srcV };
        for (int reg = 0; reg < 3; ++reg) {
            const __bf16* s = srcs[reg];
            __bf16* dstBase = ldsA + reg * REG_ELEMS;
            for (int v = tid; v < 2048; v += 256) {     // 2048 uint4 per region
                int row = v >> 4, c8 = v & 15;
                *(uint4*)(dstBase + row * REG_STRIDE + c8 * 8) =
                    *(const uint4*)(s + row * 128 + c8 * 8);
            }
        }
    }
#endif

    const int lane  = tid & 31;
    const int laneN = lane & 15;
    const int half  = lane >> 4;
    const int mt    = tid >> 5;                // 8 waves = 8 M-tiles of 16 rows
    const int mrow  = mt * 16 + laneN;

    union Frag { v16bf v; uint4 q[2]; };
    Frag aF[KTOT / 32];
    bool aLoaded = false;

    for (int ut = 0; ut < 4; ++ut) {
#if USE_TDM
        if (tid < 32) __builtin_amdgcn_s_wait_tensorcnt(0);   // B(ut) landed
        __syncthreads();                       // release waves; WAR fence too
        if (tid < 32 && ut < 3)                // stream next tile under WMMAs
            tdm_load_padded(ldsBBase + ((unsigned)((ut + 1) & 1)) * (BBUF_ELEMS * 2),
                            srcB + (size_t)(ut + 1) * (64 * KTOT), BCHUNKS);
        const __bf16* bbase = ldsB + (ut & 1) * BBUF_ELEMS;
#else
        __syncthreads();                       // previous compute done
        {
            const __bf16* s = srcB + (size_t)ut * (64 * KTOT);
            for (int v = tid; v < BCHUNKS * 16; v += 256) {   // 16 uint4 / chunk
                int ch = v >> 4, c8 = v & 15;
                *(uint4*)(ldsB + ch * REG_STRIDE + c8 * 8) =
                    *(const uint4*)(s + ch * 128 + c8 * 8);
            }
        }
        __syncthreads();
        const __bf16* bbase = ldsB;
#endif

        if (!aLoaded) {                        // preload A frags once (ut-invariant)
            #pragma unroll
            for (int kt = 0; kt < KTOT / 32; ++kt) {
                const int kb = kt * 32 + half * 8;
                const __bf16* ap = ldsA + (kb >> 7) * REG_ELEMS
                                 + mrow * REG_STRIDE + (kb & 127);
                aF[kt].q[0] = *(const uint4*)ap;
                aF[kt].q[1] = *(const uint4*)(ap + 16);
            }
            aLoaded = true;
        }

        const int unit = blkInCell * 64 + ut * 16 + laneN;    // output unit 0..255

        v8f acc0, acc1, acc2, acc3;            // i, f, g, o
        {
            float b0 = bias[0 * 256 + unit];
            float b1 = bias[1 * 256 + unit];
            float b2 = bias[2 * 256 + unit];
            float b3 = bias[3 * 256 + unit];
            #pragma unroll
            for (int e = 0; e < 8; ++e) {
                acc0[e] = b0; acc1[e] = b1; acc2[e] = b2; acc3[e] = b3;
            }
        }

        #pragma unroll
        for (int kt = 0; kt < KTOT / 32; ++kt) {
            const int kb     = kt * 32 + half * 8;
            const int region = kb >> 7;
            const int kloc   = kb & 127;
            Frag f0, f1, f2, f3;
            const __bf16* p0 = bbase + ((0 * 16 + laneN) * 3 + region) * REG_STRIDE + kloc;
            const __bf16* p1 = bbase + ((1 * 16 + laneN) * 3 + region) * REG_STRIDE + kloc;
            const __bf16* p2 = bbase + ((2 * 16 + laneN) * 3 + region) * REG_STRIDE + kloc;
            const __bf16* p3 = bbase + ((3 * 16 + laneN) * 3 + region) * REG_STRIDE + kloc;
            f0.q[0] = *(const uint4*)p0; f0.q[1] = *(const uint4*)(p0 + 16);
            f1.q[0] = *(const uint4*)p1; f1.q[1] = *(const uint4*)(p1 + 16);
            f2.q[0] = *(const uint4*)p2; f2.q[1] = *(const uint4*)(p2 + 16);
            f3.q[0] = *(const uint4*)p3; f3.q[1] = *(const uint4*)(p3 + 16);
            acc0 = __builtin_amdgcn_wmma_f32_16x16x32_bf16(false, aF[kt].v, false, f0.v, (short)0, acc0, false, false);
            acc1 = __builtin_amdgcn_wmma_f32_16x16x32_bf16(false, aF[kt].v, false, f1.v, (short)0, acc1, false, false);
            acc2 = __builtin_amdgcn_wmma_f32_16x16x32_bf16(false, aF[kt].v, false, f2.v, (short)0, acc2, false, false);
            acc3 = __builtin_amdgcn_wmma_f32_16x16x32_bf16(false, aF[kt].v, false, f3.v, (short)0, acc3, false, false);
        }

        // ---- fused LSTM elementwise in registers (D layout: m = v + half*8) -
        __bf16* hcW = hColW + r * (BATCHN * HDIM);
        float*  ccW = cColW + r * (BATCHN * HDIM);
        __bf16* hrW = hRowW + c * (BATCHN * HDIM);
        float*  crW = cRowW + c * (BATCHN * HDIM);
        const float* ccRp = cColR + r * (BATCHN * HDIM);
        const float* crRp = cRowR + c * (BATCHN * HDIM);

        #pragma unroll
        for (int v = 0; v < 8; ++v) {
            const int   b  = mt * 16 + half * 8 + v;          // batch row
            const float iv = sigmoidf_(acc0[v]);
            const float fv = sigmoidf_(acc1[v]);
            const float gv = tanhf(acc2[v]);
            const float ov = sigmoidf_(acc3[v]);
            if (unit < 128) {                                 // column-direction
                const int off = b * HDIM + unit;
                const float cp = (c == 0) ? 0.0f : ccRp[off];
                const float c2 = fv * cp + iv * gv;
                ccW[off] = c2;
                hcW[off] = (__bf16)(ov * tanhf(c2));
            } else {                                          // row-direction
                const int off = b * HDIM + (unit - 128);
                const float cp = (r == 0) ? 0.0f : crRp[off];
                const float c2 = fv * cp + iv * gv;
                crW[off] = c2;
                hrW[off] = (__bf16)(ov * tanhf(c2));
            }
        }
    }
}

// ------------------------------------------------------------- finalize -----
__global__ __launch_bounds__(256) void finalize_k(
    const __bf16* __restrict__ hcol31, const float* __restrict__ ccol31,
    const __bf16* __restrict__ hrow31, const float* __restrict__ crow31,
    float* __restrict__ out)
{
    int gid = blockIdx.x * 256 + threadIdx.x;        // < 65536
    int isC = gid >> 15;
    int rem = gid & 32767;
    int b   = rem >> 8;
    int u   = rem & 255;
    float val;
    if (isC) val = (u < 128) ? ccol31[b * 128 + u] : crow31[b * 128 + (u - 128)];
    else     val = (u < 128) ? (float)hcol31[b * 128 + u]
                             : (float)hrow31[b * 128 + (u - 128)];
    out[gid] = val;
}

// ----------------------------------------------------------------- host -----
extern "C" void kernel_launch(void* const* d_in, const int* in_sizes, int n_in,
                              void* d_out, int out_size, void* d_ws, size_t ws_size,
                              hipStream_t stream)
{
    (void)in_sizes; (void)n_in; (void)out_size; (void)ws_size;
    const float* x    = (const float*)d_in[0];
    const float* w_ih = (const float*)d_in[1];
    const float* w_hh = (const float*)d_in[2];
    const float* b_ih = (const float*)d_in[3];
    const float* b_hh = (const float*)d_in[4];

    char* ws = (char*)d_ws;
    __bf16* Wb    = (__bf16*)(ws + 0);                  //    786,432 B
    float*  bias  = (float*) (ws + 786432);             //      4,096 B
    __bf16* zeroB = (__bf16*)(ws + 790528);             //     32,768 B
    __bf16* xb    = (__bf16*)(ws + 823296);             // 33,554,432 B
    float*  Ccol  = (float*) (ws + 34377728ULL);        //  4,194,304 B (2 slots)
    float*  Crow  = (float*) (ws + 38572032ULL);        //  4,194,304 B
    __bf16* Hcol  = (__bf16*)(ws + 42766336ULL);        //  2,097,152 B
    __bf16* Hrow  = (__bf16*)(ws + 44863488ULL);        //  2,097,152 B

    prep_k<<<(32 * 32 * BATCHN * 128) / 256, 256, 0, stream>>>(
        x, w_ih, w_hh, b_ih, b_hh, Wb, bias, xb, zeroB);

    const size_t ldsBytes = (A_ELEMS + 2 * BBUF_ELEMS) * 2;   // 208,896 B
    for (int d = 0; d < 63; ++d) {
        int rmin   = (d > 31) ? (d - 31) : 0;
        int rmax   = (d < 31) ? d : 31;
        int ncells = rmax - rmin + 1;
        int wsl = d & 1, rsl = 1 - wsl;                 // parity double-buffer
        grid_lstm_diag<<<ncells * 4, 256, ldsBytes, stream>>>(
            xb, Wb, bias, zeroB,
            Hcol + rsl * SLOTE, Hcol + wsl * SLOTE,
            Ccol + rsl * SLOTE, Ccol + wsl * SLOTE,
            Hrow + rsl * SLOTE, Hrow + wsl * SLOTE,
            Crow + rsl * SLOTE, Crow + wsl * SLOTE,
            d);
    }

    // d=62 wrote parity slot 0; row/col index 31 holds the final states.
    const int ROWOFF = 31 * BATCHN * HDIM;
    finalize_k<<<65536 / 256, 256, 0, stream>>>(
        Hcol + ROWOFF, Ccol + ROWOFF, Hrow + ROWOFF, Crow + ROWOFF, (float*)d_out);
}
